// GptOssDecoderLayer_84172769068066
// MI455X (gfx1250) — compile-verified, matrix operand order checked by reading, not run on hardware
//
#include <hip/hip_runtime.h>

// ---------------- static problem config (mirrors reference) ----------------
constexpr int Hc   = 1024;   // hidden
constexpr int HDc  = 64;     // head dim
constexpr int NQc  = 16;     // query heads
constexpr int NKVc = 4;      // kv heads
constexpr int Gc   = 4;      // GQA group
constexpr int Ec   = 8;      // experts
constexpr int Fc   = 1024;   // expert FFN
constexpr int Bc   = 4;      // batch
constexpr int Qc   = 128;    // new queries / seq
constexpr int Pc   = 8;      // pages / seq
constexpr int PSc  = 128;    // page size
constexpr int Sc   = 1024;   // kv len
constexpr int Tc   = 512;    // total tokens
constexpr int SWc  = 128;    // sliding window
constexpr int SKVc = 256;    // live kv span (window-trimmed)
constexpr int SKV0 = 768;    // first live key position
constexpr int QKVW = 1536;   // HD*(NQ+2*NKV)
constexpr float LIMITc = 7.0f;
constexpr float ALPHAc = 1.702f;
constexpr float EPSc   = 1e-5f;
constexpr float SCALEc = 0.125f;   // 64^-0.5

typedef __bf16 bf16;
typedef __attribute__((ext_vector_type(16))) __bf16 v16bf;
typedef __attribute__((ext_vector_type(8)))  __bf16 v8bf;
typedef __attribute__((ext_vector_type(8)))  float  v8f;
typedef __attribute__((ext_vector_type(4)))  unsigned u32x4;
typedef __attribute__((ext_vector_type(8)))  unsigned u32x8;

__device__ __forceinline__ bf16 f2bf(float f) {
  unsigned u = __float_as_uint(f);
  u += 0x7FFFu + ((u >> 16) & 1u);           // round-to-nearest-even
  unsigned short h = (unsigned short)(u >> 16);
  return __builtin_bit_cast(bf16, h);
}

__device__ __forceinline__ v16bf cat16(v8bf lo, v8bf hi) {
  return __builtin_shufflevector(lo, hi, 0,1,2,3,4,5,6,7,8,9,10,11,12,13,14,15);
}

__device__ __forceinline__ v8f wmma_bf16(v16bf a, v16bf b, v8f c) {
  return __builtin_amdgcn_wmma_f32_16x16x32_bf16(
      false, a, false, b, (short)0, c, false, false);
}

// ---------------------------------------------------------------------------
// Tensor Data Mover: 2D tile (tile_d1 rows x tile_d0 bf16 elems) from global
// (row stride = stride0 elems) into LDS at lds_addr. Each 128B tile row is
// followed by a 16B pad in LDS (pad_interval=128B, pad_amount=16B) so
// fragment ds_load_b128 rows land on distinct banks (stride 144B = 36 DWORDs,
// gcd(36,64)=4 -> 16 lanes hit 16 distinct banks).
// ---------------------------------------------------------------------------
__device__ __forceinline__ void tdm_load_2d(unsigned lds_addr, const void* gptr,
                                            unsigned tile_d0, unsigned tile_d1,
                                            unsigned long stride0) {
  unsigned long ga = (unsigned long)gptr;
  u32x4 g0;
  g0[0] = 1u;                                        // count=1, no gather
  g0[1] = lds_addr;                                  // LDS byte address
  g0[2] = (unsigned)(ga & 0xFFFFFFFFu);              // global_addr[31:0]
  g0[3] = (unsigned)((ga >> 32) & 0x01FFFFFFu) | (2u << 30);  // addr[56:32]|type=2

  const unsigned long td0 = 1ul << 30, td1 = 1ul << 30;
  unsigned long q0 = (1ul << 16)                     // data_size = 1 (2 bytes)
                   | (1ul << 20)                     // pad_enable
                   | (4ul << 22)                     // pad_interval: 32 DW (128B)
                   | (3ul << 25)                     // pad_amount: 4 DW (16B)
                   | ((td0 & 0xFFFFul) << 48);       // tensor_dim0[15:0]
  unsigned long q1 = ((td0 >> 16) & 0xFFFFul)        // tensor_dim0[31:16]
                   | ((td1 & 0xFFFFFFFFul) << 16)    // tensor_dim1 (bits 111:80)
                   | ((unsigned long)tile_d0 << 48); // tile_dim0 (bits 127:112)
  unsigned long q2 = (unsigned long)tile_d1          // tile_dim1 (bits 143:128)
                   | ((stride0 & 0xFFFFFFFFul) << 32); // stride0[31:0] (191:160)
  unsigned long q3 = (stride0 >> 32) & 0xFFFFul;     // stride0[47:32] (207:192)

  u32x8 g1;
  g1[0] = (unsigned)q0; g1[1] = (unsigned)(q0 >> 32);
  g1[2] = (unsigned)q1; g1[3] = (unsigned)(q1 >> 32);
  g1[4] = (unsigned)q2; g1[5] = (unsigned)(q2 >> 32);
  g1[6] = (unsigned)q3; g1[7] = (unsigned)(q3 >> 32);

  asm volatile("tensor_load_to_lds %0, %1" :: "s"(g0), "s"(g1) : "memory");
}

// ---------------------------------------------------------------------------
// Batched GEMM: C[z] = A[z] (MxK bf16 row-major) * Bt[z]^T (Bt: NxK bf16)
//               + bias + resid.
// Block = 256 threads (8 waves), tile 128(M) x 64(N), K stepped by 64.
// TDM stages A(128x64) and B(64x64) panels in LDS, double-buffered so the
// DMA of tile k+1 overlaps WMMA on tile k. All fragments of a K-tile are
// loaded (clause of ds_load_b128) before the 8 back-to-back WMMAs so the
// matrix pipe sees a single dscnt wait per tile instead of one per WMMA.
// Requires M%128==0, N%64==0, K%64==0 (true for every GEMM in this layer).
// ---------------------------------------------------------------------------
constexpr int KT       = 64;
constexpr int LDS_ROWB = 144;              // 128B data + 16B TDM pad
constexpr int ASZ      = 128 * LDS_ROWB;   // 18432 B
constexpr int BSZ      = 64  * LDS_ROWB;   //  9216 B

__global__ void gemm_bf16_wmma(
    const bf16* __restrict__ A,  long sA, int lda,
    const bf16* __restrict__ Bt, long sB, int ldb,
    float* __restrict__ C, long sCb, long sCi, int zmod, int ldc,
    int M, int N, int K,
    const float* __restrict__ bias, long sBias,
    const float* __restrict__ resid)
{
  __shared__ __align__(16) unsigned char lds[2][ASZ + BSZ];

  const int z = blockIdx.z;
  const long offC = (long)(z / zmod) * sCb + (long)(z % zmod) * sCi;
  C += offC;
  if (resid) resid += offC;
  if (bias)  bias  += (long)z * sBias;

  const bf16* Ablk = A  + (long)z * sA + (long)blockIdx.x * 128 * lda;
  const bf16* Bblk = Bt + (long)z * sB + (long)blockIdx.y * 64  * ldb;

  const int wave = threadIdx.x >> 5;
  const int lane = threadIdx.x & 31;
  const int half = lane >> 4;
  const int l15  = lane & 15;
  const int m0   = blockIdx.x * 128 + wave * 16;
  const int n0   = blockIdx.y * 64;

  const unsigned ldsA0 = (unsigned)(unsigned long)(void*)&lds[0][0];
  const unsigned ldsA1 = (unsigned)(unsigned long)(void*)&lds[1][0];

  if (wave == 0) {
    tdm_load_2d(ldsA0,       Ablk, KT, 128, (unsigned long)lda);
    tdm_load_2d(ldsA0 + ASZ, Bblk, KT, 64,  (unsigned long)ldb);
  }

  v8f acc[4];
  acc[0] = v8f{}; acc[1] = v8f{}; acc[2] = v8f{}; acc[3] = v8f{};

  int buf = 0;
  for (int k0 = 0; k0 < K; k0 += KT) {
    if (wave == 0) __builtin_amdgcn_s_wait_tensorcnt(0);
    __syncthreads();                      // panel `buf` is resident
    if (wave == 0 && k0 + KT < K) {       // prefetch next panel via TDM
      unsigned dst = (buf ? ldsA0 : ldsA1);
      tdm_load_2d(dst,       Ablk + (k0 + KT), KT, 128, (unsigned long)lda);
      tdm_load_2d(dst + ASZ, Bblk + (k0 + KT), KT, 64,  (unsigned long)ldb);
    }

    const unsigned char* la = &lds[buf][0];
    const unsigned char* lb = la + ASZ;

    // ---- load ALL fragments of this K-tile first (one ds clause) ----
    v16bf afrag[2];
    v16bf bfrag[2][4];
#pragma unroll
    for (int kk2 = 0; kk2 < 2; ++kk2) {   // kk2*32 K-step, byte off = kk2*64
      const int off = kk2 * 64 + half * 16;
      const unsigned char* ar = la + (wave * 16 + l15) * LDS_ROWB + off;
      afrag[kk2] = cat16(*(const v8bf*)ar, *(const v8bf*)(ar + 32));
#pragma unroll
      for (int j = 0; j < 4; ++j) {
        const unsigned char* br = lb + (j * 16 + l15) * LDS_ROWB + off;
        bfrag[kk2][j] = cat16(*(const v8bf*)br, *(const v8bf*)(br + 32));
      }
    }
    // ---- then 8 back-to-back WMMAs ----
#pragma unroll
    for (int kk2 = 0; kk2 < 2; ++kk2)
#pragma unroll
      for (int j = 0; j < 4; ++j)
        acc[j] = wmma_bf16(afrag[kk2], bfrag[kk2][j], acc[j]);

    __syncthreads();                      // everyone done reading `buf`
    buf ^= 1;
  }

#pragma unroll
  for (int j = 0; j < 4; ++j) {
#pragma unroll
    for (int r = 0; r < 8; ++r) {
      const int m = m0 + half * 8 + r;
      const int n = n0 + j * 16 + l15;
      float v = acc[j][r];
      if (bias)  v += bias[n];
      if (resid) v += resid[(long)m * ldc + n];
      C[(long)m * ldc + n] = v;
    }
  }
}

// -------------------------- elementwise / glue -----------------------------

__global__ void transpose_cast_kernel(const float* __restrict__ in,
                                      bf16* __restrict__ out,
                                      int R, int C, long total) {
  long i = (long)blockIdx.x * blockDim.x + threadIdx.x;
  if (i >= total) return;
  long rc = (long)R * C;
  long e = i / rc, rem = i - e * rc;
  int r = (int)(rem / C), c = (int)(rem - (long)r * C);
  out[e * rc + (long)c * R + r] = f2bf(in[i]);
}

__global__ void cast_bf16_kernel(const float* __restrict__ in,
                                 bf16* __restrict__ out, long n) {
  long i = (long)blockIdx.x * blockDim.x + threadIdx.x;
  if (i < n) out[i] = f2bf(in[i]);
}

__global__ void rmsnorm_kernel(const float* __restrict__ x,
                               const float* __restrict__ w,
                               bf16* __restrict__ xb,
                               float* __restrict__ xf) {
  const int t = blockIdx.x;
  const float* row = x + (long)t * Hc;
  __shared__ float red[256];
  float s = 0.f;
  for (int i = threadIdx.x; i < Hc; i += 256) { float v = row[i]; s += v * v; }
  red[threadIdx.x] = s; __syncthreads();
  for (int st = 128; st > 0; st >>= 1) {
    if (threadIdx.x < st) red[threadIdx.x] += red[threadIdx.x + st];
    __syncthreads();
  }
  const float rs = rsqrtf(red[0] / (float)Hc + EPSc);
  for (int i = threadIdx.x; i < Hc; i += 256) {
    float v = row[i] * w[i] * rs;
    xf[(long)t * Hc + i] = v;
    xb[(long)t * Hc + i] = f2bf(v);
  }
}

__global__ void qb_kernel(const float* __restrict__ qkv, bf16* __restrict__ qb) {
  int i = blockIdx.x * 256 + threadIdx.x;
  if (i >= Tc * NQc * HDc) return;
  int d = i % HDc; int tmp = i / HDc;
  int h = tmp % NQc; int t = tmp / NQc;
  int b = t / Qc, q = t % Qc;
  float v = qkv[(long)t * QKVW + h * HDc + d] * SCALEc;
  qb[(((long)(b * NQc + h) * Qc + q) * HDc + d)] = f2bf(v);
}

__global__ void build_kv_kernel(const float* __restrict__ qkv,
                                const float* __restrict__ cache,
                                const int* __restrict__ pageidx,
                                bf16* __restrict__ Krep,
                                bf16* __restrict__ Vtrep) {
  int i = blockIdx.x * 256 + threadIdx.x;
  if (i >= Bc * NKVc * SKVc * HDc) return;
  int d  = i % HDc;  int t1 = i / HDc;
  int s1 = t1 % SKVc; int t2 = t1 / SKVc;
  int kvh = t2 % NKVc; int b = t2 / NKVc;
  int s = SKV0 + s1;
  float kval, vval;
  if (s >= Sc - Qc) {
    int t = b * Qc + (s - (Sc - Qc));
    kval = qkv[(long)t * QKVW + NQc * HDc + kvh * HDc + d];
    vval = qkv[(long)t * QKVW + (NQc + NKVc) * HDc + kvh * HDc + d];
  } else {
    int page = s / PSc, slot = s % PSc;
    int pidx = pageidx[b * Pc + page];
    long base = (long)pidx * 2 * PSc * NKVc * HDc;
    kval = cache[base + (long)slot * NKVc * HDc + kvh * HDc + d];
    vval = cache[base + (long)PSc * NKVc * HDc + (long)slot * NKVc * HDc + kvh * HDc + d];
  }
  bf16 kb = f2bf(kval), vb = f2bf(vval);
#pragma unroll
  for (int g = 0; g < Gc; ++g) {
    int h = kvh * Gc + g;
    Krep [(((long)(b * NQc + h) * SKVc) + s1) * HDc + d] = kb;
    Vtrep[(((long)(b * NQc + h) * HDc) + d) * SKVc + s1] = vb;
  }
}

__global__ void softmax_kernel(const float* __restrict__ scores,
                               const float* __restrict__ sinks,
                               bf16* __restrict__ pb) {
  const int wave = threadIdx.x >> 5, lane = threadIdx.x & 31;
  const int row = blockIdx.x * 8 + wave;      // rows = B*NQ*Q
  const int q = row % Qc;
  const int h = (row / Qc) % NQc;
  const int qpos = Sc - Qc + q;
  const float* srow = scores + (long)row * SKVc;
  float vals[8];
  float m = -1e30f;
#pragma unroll
  for (int i = 0; i < 8; ++i) {
    int s1 = i * 32 + lane;
    int kpos = SKV0 + s1;
    bool ok = (qpos >= kpos) && ((qpos - kpos) < SWc);
    float v = ok ? srow[s1] : -1e30f;
    vals[i] = v;
    m = fmaxf(m, v);
  }
  for (int o = 16; o > 0; o >>= 1) m = fmaxf(m, __shfl_xor(m, o, 32));
  float sum = 0.f;
#pragma unroll
  for (int i = 0; i < 8; ++i) {
    float p = (vals[i] <= -1e29f) ? 0.f : __expf(vals[i] - m);
    vals[i] = p; sum += p;
  }
  for (int o = 16; o > 0; o >>= 1) sum += __shfl_xor(sum, o, 32);
  sum += __expf(sinks[h] - m);
  const float inv = 1.0f / sum;
#pragma unroll
  for (int i = 0; i < 8; ++i) {
    int s1 = i * 32 + lane;
    pb[(long)row * SKVc + s1] = f2bf(vals[i] * inv);
  }
}

__global__ void router_kernel(const float* __restrict__ x2,
                              const float* __restrict__ wr,
                              const float* __restrict__ br,
                              float* __restrict__ comb) {
  const int wave = threadIdx.x >> 5, lane = threadIdx.x & 31;
  const int t = blockIdx.x * 8 + wave;
  float acc[Ec];
#pragma unroll
  for (int e = 0; e < Ec; ++e) acc[e] = 0.f;
  for (int hh = lane; hh < Hc; hh += 32) {
    float xv = x2[(long)t * Hc + hh];
#pragma unroll
    for (int e = 0; e < Ec; ++e) acc[e] += xv * wr[(long)hh * Ec + e];
  }
#pragma unroll
  for (int e = 0; e < Ec; ++e)
    for (int o = 16; o > 0; o >>= 1) acc[e] += __shfl_xor(acc[e], o, 32);
  if (lane == 0) {
    float lg[Ec];
#pragma unroll
    for (int e = 0; e < Ec; ++e) lg[e] = acc[e] + br[e];
    int i0 = 0;
#pragma unroll
    for (int e = 1; e < Ec; ++e) if (lg[e] > lg[i0]) i0 = e;
    int i1 = (i0 == 0) ? 1 : 0;
#pragma unroll
    for (int e = 0; e < Ec; ++e) if (e != i0 && lg[e] > lg[i1]) i1 = e;
    float e1 = __expf(lg[i1] - lg[i0]);
    float den = 1.0f + e1;
#pragma unroll
    for (int e = 0; e < Ec; ++e) comb[t * Ec + e] = 0.f;
    comb[t * Ec + i0] = 1.0f / den;
    comb[t * Ec + i1] = e1 / den;
  }
}

__global__ void swiglu_kernel(const float* __restrict__ gu,
                              bf16* __restrict__ act, long total) {
  long i = (long)blockIdx.x * 256 + threadIdx.x;
  if (i >= total) return;
  long f = i % Fc;
  long base2 = (i / Fc) * (2 * (long)Fc);
  float g = gu[base2 + 2 * f];
  float u = gu[base2 + 2 * f + 1];
  g = fminf(g, LIMITc);
  u = fminf(fmaxf(u, -LIMITc), LIMITc);
  float glu = g / (1.0f + __expf(-ALPHAc * g));
  act[i] = f2bf((u + 1.0f) * glu);
}

__global__ void combine_kernel(const float* __restrict__ hbuf,
                               const float* __restrict__ eo,
                               const float* __restrict__ comb,
                               float* __restrict__ out) {
  long i = (long)blockIdx.x * 256 + threadIdx.x;
  if (i >= (long)Tc * Hc) return;
  int t = (int)(i / Hc);
  float s = hbuf[i];
#pragma unroll
  for (int e = 0; e < Ec; ++e)
    s += comb[t * Ec + e] * eo[(long)e * Tc * Hc + i];
  out[i] = s;
}

// ---------------------------------------------------------------------------

extern "C" void kernel_launch(void* const* d_in, const int* in_sizes, int n_in,
                              void* d_out, int out_size, void* d_ws, size_t ws_size,
                              hipStream_t stream) {
  const float* hidden   = (const float*)d_in[0];
  const float* kv_cache = (const float*)d_in[1];
  const int*   pageidx  = (const int*)d_in[3];
  const float* sinks    = (const float*)d_in[4];
  const float* w_qkv    = (const float*)d_in[5];
  const float* b_qkv    = (const float*)d_in[6];
  const float* w_o      = (const float*)d_in[7];
  const float* b_o      = (const float*)d_in[8];
  const float* ln1_w    = (const float*)d_in[9];
  const float* ln2_w    = (const float*)d_in[10];
  const float* w_router = (const float*)d_in[11];
  const float* b_router = (const float*)d_in[12];
  const float* w_gate_up= (const float*)d_in[13];
  const float* b_gate_up= (const float*)d_in[14];
  const float* w_down   = (const float*)d_in[15];
  const float* b_down   = (const float*)d_in[16];
  float* out = (float*)d_out;

  char* p = (char*)d_ws;
  auto alloc = [&](size_t bytes) -> void* {
    void* r = (void*)p;
    p += (bytes + 255) & ~(size_t)255;
    return r;
  };
  bf16*  wqkv_t = (bf16*) alloc((size_t)QKVW * Hc * 2);
  bf16*  wo_t   = (bf16*) alloc((size_t)Hc * Hc * 2);
  bf16*  wgu_t  = (bf16*) alloc((size_t)Ec * 2 * Fc * Hc * 2);
  bf16*  wdn_t  = (bf16*) alloc((size_t)Ec * Hc * Fc * 2);
  bf16*  xb     = (bf16*) alloc((size_t)Tc * Hc * 2);
  bf16*  x2b    = (bf16*) alloc((size_t)Tc * Hc * 2);
  float* normf  = (float*)alloc((size_t)Tc * Hc * 4);
  float* qkv    = (float*)alloc((size_t)Tc * QKVW * 4);
  bf16*  qb     = (bf16*) alloc((size_t)Bc * NQc * Qc * HDc * 2);
  bf16*  Krep   = (bf16*) alloc((size_t)Bc * NQc * SKVc * HDc * 2);
  bf16*  Vtrep  = (bf16*) alloc((size_t)Bc * NQc * HDc * SKVc * 2);
  float* scores = (float*)alloc((size_t)Bc * NQc * Qc * SKVc * 4);
  bf16*  pb     = (bf16*) alloc((size_t)Bc * NQc * Qc * SKVc * 2);
  float* attn   = (float*)alloc((size_t)Tc * NQc * HDc * 4);
  bf16*  attnb  = (bf16*) alloc((size_t)Tc * NQc * HDc * 2);
  float* hbuf   = (float*)alloc((size_t)Tc * Hc * 4);
  float* comb   = (float*)alloc((size_t)Tc * Ec * 4);
  float* gu     = (float*)alloc((size_t)Ec * Tc * 2 * Fc * 4);
  bf16*  act    = (bf16*) alloc((size_t)Ec * Tc * Fc * 2);
  float* eo     = (float*)alloc((size_t)Ec * Tc * Hc * 4);
  (void)ws_size; (void)in_sizes; (void)n_in; (void)out_size;

  auto gemm = [&](const bf16* A, long sA, int lda,
                  const bf16* Bt, long sB, int ldb,
                  float* C, long sCb, long sCi, int zmod, int ldc,
                  int M, int N, int K,
                  const float* bias, long sBias,
                  const float* resid, int zcount) {
    dim3 grid(M / 128, N / 64, zcount);
    gemm_bf16_wmma<<<grid, 256, 0, stream>>>(A, sA, lda, Bt, sB, ldb,
                                             C, sCb, sCi, zmod, ldc,
                                             M, N, K, bias, sBias, resid);
  };
  auto ew = [&](long total) { return dim3((unsigned)((total + 255) / 256)); };

  // ---- weight transpose+cast ----
  transpose_cast_kernel<<<ew((long)Hc * QKVW), 256, 0, stream>>>(w_qkv, wqkv_t, Hc, QKVW, (long)Hc * QKVW);
  transpose_cast_kernel<<<ew((long)Hc * Hc), 256, 0, stream>>>(w_o, wo_t, Hc, Hc, (long)Hc * Hc);
  transpose_cast_kernel<<<ew((long)Ec * Hc * 2 * Fc), 256, 0, stream>>>(w_gate_up, wgu_t, Hc, 2 * Fc, (long)Ec * Hc * 2 * Fc);
  transpose_cast_kernel<<<ew((long)Ec * Fc * Hc), 256, 0, stream>>>(w_down, wdn_t, Fc, Hc, (long)Ec * Fc * Hc);

  // ---- attention ----
  rmsnorm_kernel<<<Tc, 256, 0, stream>>>(hidden, ln1_w, xb, normf);
  gemm(xb, 0, Hc, wqkv_t, 0, Hc, qkv, 0, 0, 1, QKVW,
       Tc, QKVW, Hc, b_qkv, 0, nullptr, 1);
  qb_kernel<<<ew((long)Tc * NQc * HDc), 256, 0, stream>>>(qkv, qb);
  build_kv_kernel<<<ew((long)Bc * NKVc * SKVc * HDc), 256, 0, stream>>>(qkv, kv_cache, pageidx, Krep, Vtrep);
  gemm(qb, (long)Qc * HDc, HDc, Krep, (long)SKVc * HDc, HDc,
       scores, (long)Qc * SKVc, 0, 1, SKVc,
       Qc, SKVc, HDc, nullptr, 0, nullptr, Bc * NQc);
  softmax_kernel<<<(Bc * NQc * Qc) / 8, 256, 0, stream>>>(scores, sinks, pb);
  gemm(pb, (long)Qc * SKVc, SKVc, Vtrep, (long)HDc * SKVc, SKVc,
       attn, (long)Qc * NQc * HDc, HDc, NQc, NQc * HDc,
       Qc, HDc, SKVc, nullptr, 0, nullptr, Bc * NQc);
  cast_bf16_kernel<<<ew((long)Tc * NQc * HDc), 256, 0, stream>>>(attn, attnb, (long)Tc * NQc * HDc);
  gemm(attnb, 0, NQc * HDc, wo_t, 0, NQc * HDc, hbuf, 0, 0, 1, Hc,
       Tc, Hc, NQc * HDc, b_o, 0, hidden, 1);

  // ---- MoE ----
  rmsnorm_kernel<<<Tc, 256, 0, stream>>>(hbuf, ln2_w, x2b, normf);
  router_kernel<<<Tc / 8, 256, 0, stream>>>(normf, w_router, b_router, comb);
  gemm(x2b, 0, Hc, wgu_t, (long)2 * Fc * Hc, Hc,
       gu, (long)Tc * 2 * Fc, 0, 1, 2 * Fc,
       Tc, 2 * Fc, Hc, b_gate_up, 2 * Fc, nullptr, Ec);
  swiglu_kernel<<<ew((long)Ec * Tc * Fc), 256, 0, stream>>>(gu, act, (long)Ec * Tc * Fc);
  gemm(act, (long)Tc * Fc, Fc, wdn_t, (long)Hc * Fc, Fc,
       eo, (long)Tc * Hc, 0, 1, Hc,
       Tc, Hc, Fc, b_down, Hc, nullptr, Ec);
  combine_kernel<<<ew((long)Tc * Hc), 256, 0, stream>>>(hbuf, eo, comb, out);
}
